// rnnEncoder_29368986370321
// MI455X (gfx1250) — matrix operand order repaired and checked by reading, not compiled, not run on hardware
//
#include <hip/hip_runtime.h>

// Problem dims (fixed by the reference)
#define T_STEPS 4096
#define EMB     2048
#define OUT     2048
#define GATES   (4 * OUT)   // 8192

typedef __attribute__((ext_vector_type(16))) _Float16 v16h;
typedef __attribute__((ext_vector_type(8)))  _Float16 v8h;
typedef __attribute__((ext_vector_type(8)))  float    v8f;

union FragF16 { v16h h; uint4 q[2]; };

// ---------------- Workspace layout (bytes, 256-aligned) ----------------
#define XPROJ_OFF  ((size_t)0)                                   // 4096*8192*4 = 128 MiB
#define A16_OFF    (XPROJ_OFF + (size_t)T_STEPS * GATES * 4)     // 4096*2048*2 = 16 MiB
#define WIH16_OFF  (A16_OFF   + (size_t)T_STEPS * EMB   * 2)     // 8192*2048*2 = 32 MiB
#define WHH16_OFF  (WIH16_OFF + (size_t)GATES   * EMB   * 2)     // 8192*2048*2 = 32 MiB
#define BIAS_OFF   (WHH16_OFF + (size_t)GATES   * OUT   * 2)     // 8192*4
#define HBUF_OFF   (BIAS_OFF  + (size_t)GATES * 4)               // 2*2048*4
#define CNT_OFF    (HBUF_OFF  + (size_t)2 * OUT * 4)             // 4 bytes

// =======================================================================
// Prep: f32 -> f16 conversions, bias fold, h ping-pong init, barrier reset
// =======================================================================
__global__ void lstm_prep(const float* __restrict__ A,
                          const float* __restrict__ Wih,
                          const float* __restrict__ Whh,
                          const float* __restrict__ b_ih,
                          const float* __restrict__ b_hh,
                          const float* __restrict__ init_hid,
                          _Float16* __restrict__ A16,
                          _Float16* __restrict__ Wih16,
                          _Float16* __restrict__ Whh16,
                          float* __restrict__ bias_sum,
                          float* __restrict__ h_buf,
                          unsigned int* __restrict__ counter)
{
    const size_t tid    = (size_t)blockIdx.x * blockDim.x + threadIdx.x;
    const size_t stride = (size_t)gridDim.x * blockDim.x;
    for (size_t k = tid; k < (size_t)T_STEPS * EMB; k += stride) A16[k]   = (_Float16)A[k];
    for (size_t k = tid; k < (size_t)GATES * EMB;   k += stride) Wih16[k] = (_Float16)Wih[k];
    for (size_t k = tid; k < (size_t)GATES * OUT;   k += stride) Whh16[k] = (_Float16)Whh[k];
    for (size_t k = tid; k < (size_t)GATES;         k += stride) bias_sum[k] = b_ih[k] + b_hh[k];
    for (size_t k = tid; k < (size_t)OUT;           k += stride) h_buf[k] = init_hid[k];
    if (tid == 0) *counter = 0u;   // fresh barrier state every launch (graph-replay safe)
}

// =======================================================================
// Phase 1: x_proj[T, 4*OUT] = A @ W_ih^T + (b_ih + b_hh), f16 WMMA, f32 acc
// One wave32 per 16x16 output tile; K-loop 2048/32 = 64 WMMAs per wave.
// =======================================================================
__global__ void __launch_bounds__(256)
lstm_xproj_wmma(const _Float16* __restrict__ A,     // [T_STEPS, EMB] row-major
                const _Float16* __restrict__ W,     // [GATES, EMB] row-major (row n = B column n)
                const float*    __restrict__ bias,  // [GATES]
                float*          __restrict__ xproj) // [T_STEPS, GATES]
{
    const int wave  = (int)((blockIdx.x * blockDim.x + threadIdx.x) >> 5);
    const int lane  = (int)(threadIdx.x & 31);
    const int nTN   = GATES / 16;                 // 512 N-tiles
    const int tileM = (wave / nTN) * 16;
    const int tileN = (wave % nTN) * 16;
    const int r     = lane & 15;                  // row (A) / col (B) / N (C)
    const int kh    = lane >> 4;                  // K-half select

    const _Float16* aRow = A + (size_t)(tileM + r) * EMB;
    const _Float16* bRow = W + (size_t)(tileN + r) * EMB;

    // C/D layout: element (vgpr v, lane L) = (M = v + 8*(L>>4), N = L&15);
    // bias depends on N only -> same value in all 8 acc regs of a lane.
    const float bval = bias[tileN + r];
    v8f acc;
#pragma unroll
    for (int v = 0; v < 8; ++v) acc[v] = bval;

    for (int kb = 0; kb < EMB; kb += 32) {
        FragF16 fa, fb;
        // A 16x32 f16 layout: lane group kh: v0..3 = K[kb+8*kh .. +8), v4..7 = K[kb+16+8*kh .. +8)
        fa.q[0] = *(const uint4*)(aRow + kb + 8 * kh);
        fa.q[1] = *(const uint4*)(aRow + kb + 16 + 8 * kh);
        // B 32x16 f16 layout: lane group kh: v0..3 = K[kb+8*kh .. +8), v4..7 = K[kb+16+8*kh .. +8) of column (tileN + r)
        fb.q[0] = *(const uint4*)(bRow + kb + 8 * kh);
        fb.q[1] = *(const uint4*)(bRow + kb + 16 + 8 * kh);

        acc = __builtin_amdgcn_wmma_f32_16x16x32_f16(
                  /*neg_a=*/false, fa.h, /*neg_b=*/false, fb.h,
                  /*c_mod=*/(short)0, acc, /*reuse_a=*/false, /*reuse_b=*/false);
    }

#pragma unroll
    for (int v = 0; v < 8; ++v)
        xproj[(size_t)(tileM + v + 8 * kh) * GATES + tileN + r] = acc[v];
}

// =======================================================================
// Phase 2: persistent LSTM recurrence. 64 blocks x 512 threads.
// Block owns 32 cells; thread (cl, kp) does 4 gate partial dots over a
// 128-wide K segment. c lives in owner-thread registers across all steps.
// Grid barrier: monotonic atomic counter, one barrier/step, h ping-pong.
// =======================================================================
#define SEQ_BLOCKS 64
#define SEQ_THREADS 512
#define KPARTS 16                   // 512 / 32
#define KSEG   (OUT / KPARTS)       // 128
#define CPB    32                   // cells per block

__global__ void __launch_bounds__(SEQ_THREADS)
lstm_seq(const _Float16* __restrict__ Whh16,   // [GATES, OUT] f16
         const float*    __restrict__ xproj,   // [T_STEPS, GATES]
         const float*    __restrict__ init_cell,
         float*          __restrict__ h_buf,   // [2][OUT] ping-pong
         unsigned int*   __restrict__ counter,
         float*          __restrict__ out)     // [2*OUT]: h then c
{
    __shared__ float h_lds[OUT];
    __shared__ float red[CPB * 4 * KPARTS];
    __shared__ float gsum[CPB * 4];

    const int tid = (int)threadIdx.x;
    const int cl  = tid & 31;        // cell within block
    const int kp  = tid >> 5;        // K partition 0..15
    const int j   = (int)blockIdx.x * CPB + cl;   // global cell 0..2047

    // PyTorch gate order i,f,g,o -> rows j, OUT+j, 2*OUT+j, 3*OUT+j of W_hh
    const _Float16* w0 = Whh16 + ((size_t)(0 * OUT + j)) * OUT + kp * KSEG;
    const _Float16* w1 = Whh16 + ((size_t)(1 * OUT + j)) * OUT + kp * KSEG;
    const _Float16* w2 = Whh16 + ((size_t)(2 * OUT + j)) * OUT + kp * KSEG;
    const _Float16* w3 = Whh16 + ((size_t)(3 * OUT + j)) * OUT + kp * KSEG;

    float c = (kp == 0) ? init_cell[j] : 0.0f;

    for (int t = 0; t < T_STEPS; ++t) {
        // ---- stage h (device-coherent loads: written by other blocks) ----
        const float* hsrc = h_buf + (t & 1) * OUT;
#pragma unroll
        for (int u = 0; u < OUT / SEQ_THREADS; ++u) {
            const int idx = tid + u * SEQ_THREADS;
            h_lds[idx] = __hip_atomic_load(&hsrc[idx], __ATOMIC_RELAXED,
                                           __HIP_MEMORY_SCOPE_AGENT);
        }
        __syncthreads();

        // ---- partial dots: 4 gates x 128 K, f16 weights, f32 accumulate ----
        float a0 = 0.f, a1 = 0.f, a2 = 0.f, a3 = 0.f;
        const float* hseg = h_lds + kp * KSEG;
#pragma unroll 4
        for (int kk = 0; kk < KSEG; kk += 8) {
            v8h q0 = *(const v8h*)(w0 + kk);
            v8h q1 = *(const v8h*)(w1 + kk);
            v8h q2 = *(const v8h*)(w2 + kk);
            v8h q3 = *(const v8h*)(w3 + kk);
#pragma unroll
            for (int u = 0; u < 8; ++u) {
                const float hv = hseg[kk + u];
                a0 = __builtin_fmaf((float)q0[u], hv, a0);
                a1 = __builtin_fmaf((float)q1[u], hv, a1);
                a2 = __builtin_fmaf((float)q2[u], hv, a2);
                a3 = __builtin_fmaf((float)q3[u], hv, a3);
            }
        }
        red[(cl * 4 + 0) * KPARTS + kp] = a0;
        red[(cl * 4 + 1) * KPARTS + kp] = a1;
        red[(cl * 4 + 2) * KPARTS + kp] = a2;
        red[(cl * 4 + 3) * KPARTS + kp] = a3;
        __syncthreads();

        // ---- reduce 16 partials per (cell,gate): 128 tasks ----
        if (tid < CPB * 4) {
            const float* p = red + tid * KPARTS;
            float s = 0.f;
#pragma unroll
            for (int u = 0; u < KPARTS; ++u) s += p[u];
            gsum[tid] = s;
        }
        __syncthreads();

        // ---- cell update by 32 owner threads (kp == 0) ----
        if (kp == 0) {
            const float* xp = xproj + (size_t)t * GATES;
            float gi = gsum[cl * 4 + 0] + xp[0 * OUT + j];
            float gf = gsum[cl * 4 + 1] + xp[1 * OUT + j];
            float gg = gsum[cl * 4 + 2] + xp[2 * OUT + j];
            float go = gsum[cl * 4 + 3] + xp[3 * OUT + j];
            gi = 1.0f / (1.0f + __expf(-gi));
            gf = 1.0f / (1.0f + __expf(-gf));
            gg = tanhf(gg);
            go = 1.0f / (1.0f + __expf(-go));
            c = gf * c + gi * gg;
            const float h = go * tanhf(c);
            __hip_atomic_store(&h_buf[((t + 1) & 1) * OUT + j], h,
                               __ATOMIC_RELAXED, __HIP_MEMORY_SCOPE_AGENT);
            if (t == T_STEPS - 1) { out[j] = h; out[OUT + j] = c; }
        }
        __syncthreads();

        // ---- grid barrier (monotonic counter; 64 resident blocks, no deadlock) ----
        if (tid == 0) {
            __threadfence();
            __hip_atomic_fetch_add(counter, 1u, __ATOMIC_ACQ_REL,
                                   __HIP_MEMORY_SCOPE_AGENT);
            const unsigned int target = (unsigned int)(t + 1) * SEQ_BLOCKS;
            while (__hip_atomic_load(counter, __ATOMIC_ACQUIRE,
                                     __HIP_MEMORY_SCOPE_AGENT) < target)
                __builtin_amdgcn_s_sleep(1);
            __threadfence();
        }
        __syncthreads();
    }
}

// =======================================================================
extern "C" void kernel_launch(void* const* d_in, const int* in_sizes, int n_in,
                              void* d_out, int out_size, void* d_ws, size_t ws_size,
                              hipStream_t stream)
{
    const float* stacked   = (const float*)d_in[0];  // [T, EMB]
    const float* init_hid  = (const float*)d_in[1];  // [OUT]
    const float* init_cell = (const float*)d_in[2];  // [OUT]
    const float* W_ih      = (const float*)d_in[3];  // [4*OUT, EMB]
    const float* W_hh      = (const float*)d_in[4];  // [4*OUT, OUT]
    const float* b_ih      = (const float*)d_in[5];  // [4*OUT]
    const float* b_hh      = (const float*)d_in[6];  // [4*OUT]

    char* ws = (char*)d_ws;
    float*        xproj    = (float*)(ws + XPROJ_OFF);
    _Float16*     A16      = (_Float16*)(ws + A16_OFF);
    _Float16*     Wih16    = (_Float16*)(ws + WIH16_OFF);
    _Float16*     Whh16    = (_Float16*)(ws + WHH16_OFF);
    float*        bias_sum = (float*)(ws + BIAS_OFF);
    float*        h_buf    = (float*)(ws + HBUF_OFF);
    unsigned int* counter  = (unsigned int*)(ws + CNT_OFF);

    lstm_prep<<<2048, 256, 0, stream>>>(stacked, W_ih, W_hh, b_ih, b_hh, init_hid,
                                        A16, Wih16, Whh16, bias_sum, h_buf, counter);

    // 256 M-tiles * 512 N-tiles = 131072 waves -> 16384 blocks of 8 waves
    lstm_xproj_wmma<<<16384, 256, 0, stream>>>(A16, Wih16, bias_sum, xproj);

    lstm_seq<<<SEQ_BLOCKS, SEQ_THREADS, 0, stream>>>(Whh16, xproj, init_cell,
                                                     h_buf, counter, (float*)d_out);
}
